// GazePredictionModelWithDecoder_7318624272907
// MI455X (gfx1250) — matrix operand order repaired and checked by reading, not compiled, not run on hardware
//
#include <hip/hip_runtime.h>
#include <hip/hip_bf16.h>
#include <math.h>

// ---------------------------------------------------------------------------
// MI455X (gfx1250) implementation.
//  * All GEMMs lowered to v_wmma_f32_16x16x32_f16 (f16 in, f32 acc).
//  * Weights packed once per launch into WMMA B-fragment-major layout so each
//    lane's B operand is one aligned 32B vector load.
//  * LSTM input-projection (K=17, zero-padded to 32) fused into the recurrent
//    GEMM as chunk 0 of a combined [544,2048] weight -> no giant xz buffer.
//  * Recurrent step: block = 4 waves sharing one N-tile; weight fragments are
//    staged through LDS with GLOBAL_LOAD_ASYNC_TO_LDS_B128 (ASYNCcnt path)
//    to cut redundant L2 weight traffic 4x.
//  * 256 step launches (fwd+bwd fused) handle the sequential T dependency;
//    graph-capture friendly (all on `stream`, no allocs/syncs).
// ---------------------------------------------------------------------------

typedef __attribute__((ext_vector_type(16))) _Float16 v16h;
typedef __attribute__((ext_vector_type(8)))  _Float16 v8h;
typedef __attribute__((ext_vector_type(8)))  float    v8f;
typedef __attribute__((__vector_size__(4 * sizeof(int)))) int v4i_t;

#define BB 256
#define TT 256
#define FF 512
#define HH 512
#define EPSF 1e-3f

#if defined(__has_builtin)
#if __has_builtin(__builtin_amdgcn_global_load_async_to_lds_b128)
#define ASYNC_LDS 1
#endif
#endif

__device__ __forceinline__ float sigm(float x) { return 1.0f / (1.0f + __expf(-x)); }

__device__ __forceinline__ void wait_async_and_bar() {
#ifdef ASYNC_LDS
#if __has_builtin(__builtin_amdgcn_s_wait_asynccnt)
  __builtin_amdgcn_s_wait_asynccnt(0);
#else
  asm volatile("s_wait_asynccnt 0" ::: "memory");
#endif
#endif
  __syncthreads();
}

// Copy 16 bytes (8 halves) global -> LDS.
__device__ __forceinline__ void stage16(const _Float16* __restrict__ src,
                                        _Float16* __restrict__ dst_lds) {
#ifdef ASYNC_LDS
  __builtin_amdgcn_global_load_async_to_lds_b128(
      (__attribute__((address_space(1))) v4i_t*)src,
      (__attribute__((address_space(3))) v4i_t*)dst_lds, 0, 0);
#else
  *(v8h*)dst_lds = *(const v8h*)src;
#endif
}

// A-fragment (16x32 f16, row-major src, stride lda). Contiguous-k layout:
// lane L: row = L&15; halves 0..15 = K[kh..kh+15],
// kh = (L>>4)*16.
__device__ __forceinline__ v16h load_a16(const _Float16* A, int lda) {
  const int l   = threadIdx.x & 31;
  const int row = l & 15;
  const int kh  = (l >> 4) * 16;
  const _Float16* p = A + row * lda;
  v16h a;
#pragma unroll
  for (int i = 0; i < 8; ++i) { a[i] = p[kh + i]; a[8 + i] = p[kh + 8 + i]; }
  return a;
}

__device__ __forceinline__ v16h load_a16_f32(const float* A, int lda) {
  const int l   = threadIdx.x & 31;
  const int row = l & 15;
  const int kh  = (l >> 4) * 16;
  const float* p = A + row * lda;
  v16h a;
#pragma unroll
  for (int i = 0; i < 8; ++i) {
    a[i]     = (_Float16)p[kh + i];
    a[8 + i] = (_Float16)p[kh + 8 + i];
  }
  return a;
}

// B-fragment from pre-swizzled weight: fragment (kc,nt) is 512 contiguous
// halves, lane l owns halves [l*16 .. l*16+15] (32B aligned vector load).
__device__ __forceinline__ v16h load_b_swz(const _Float16* fragp) {
  return *(const v16h*)(fragp + ((threadIdx.x & 31) << 4));
}

__device__ __forceinline__ v8f wmma_f16(v16h a, v16h b, v8f c) {
  return __builtin_amdgcn_wmma_f32_16x16x32_f16(false, a, false, b, (short)0, c,
                                                false, false);
}

// ---------------------------------------------------------------------------
// Weight packing: src f32 [K,N] row-major -> swizzled B-fragment layout.
// dst element e: frag=e>>9, lane=(e>>4)&31, i=e&15;
// kc=frag/ntiles, nt=frag%ntiles, col=lane&15, kb=(lane>>4)*16;
// value = src[(kc*32+kb+i)*N + nt*16+col].
// ---------------------------------------------------------------------------
__global__ void k_pack(_Float16* __restrict__ dst, const float* __restrict__ src,
                       int K, int N) {
  int e = blockIdx.x * blockDim.x + threadIdx.x;
  int n = K * N;
  if (e >= n) return;
  int ntiles = N >> 4;
  int frag = e >> 9, lane = (e >> 4) & 31, i = e & 15;
  int kc = frag / ntiles, nt = frag - kc * ntiles;
  int col = lane & 15, kb = (lane >> 4) * 16;
  dst[e] = (_Float16)src[(size_t)(kc * 32 + kb + i) * N + nt * 16 + col];
}

// LSTM combined weight [544,2048]: rows 0..16 = wx (17 rows), 17..31 = 0,
// 32..543 = wh; packed into swizzled layout.
__global__ void k_pack_lstm(_Float16* __restrict__ dst,
                            const float* __restrict__ wx,
                            const float* __restrict__ wh) {
  int e = blockIdx.x * blockDim.x + threadIdx.x;
  const int N = 4 * HH, K = 544;
  if (e >= K * N) return;
  int ntiles = N >> 4;                       // 128
  int frag = e >> 9, lane = (e >> 4) & 31, i = e & 15;
  int kc = frag / ntiles, nt = frag - kc * ntiles;
  int col = nt * 16 + (lane & 15);
  int k = kc * 32 + (lane >> 4) * 16 + i;    // combined row 0..543
  float v;
  if (k < 17)       v = wx[(size_t)k * N + col];
  else if (k < 32)  v = 0.0f;
  else              v = wh[(size_t)(k - 32) * N + col];
  dst[e] = (_Float16)v;
}

// Decoder combined weight [1024,2048]: rows 0..511 = dec_wx, 512..1023 = dec_wh.
__global__ void k_pack_dec(_Float16* __restrict__ dst,
                           const float* __restrict__ wx,
                           const float* __restrict__ wh) {
  int e = blockIdx.x * blockDim.x + threadIdx.x;
  const int N = 4 * HH, K = 1024;
  if (e >= K * N) return;
  int ntiles = N >> 4;
  int frag = e >> 9, lane = (e >> 4) & 31, i = e & 15;
  int kc = frag / ntiles, nt = frag - kc * ntiles;
  int col = nt * 16 + (lane & 15);
  int k = kc * 32 + (lane >> 4) * 16 + i;
  float v = (k < 512) ? wx[(size_t)k * N + col] : wh[(size_t)(k - 512) * N + col];
  dst[e] = (_Float16)v;
}

__global__ void k_zero_h(_Float16* dst, int n) {
  int i = blockIdx.x * blockDim.x + threadIdx.x;
  if (i < n) dst[i] = (_Float16)0.0f;
}
__global__ void k_zero_f(float* dst, int n) {
  int i = blockIdx.x * blockDim.x + threadIdx.x;
  if (i < n) dst[i] = 0.0f;
}

// ---------------------------------------------------------------------------
// Fused LayerNorm + 4-layer MLP funnel + BN1/BN2 + seq assembly.
// Block = 128 threads (4 waves), owns 16 rows of x [B*T, 512].
// Produces seq f16 [B*T, 32]: cols 0..15 = MLP out, col 16 = y_past, 17..31 = 0.
// ---------------------------------------------------------------------------
__global__ __launch_bounds__(128) void k_mlp(
    const float* __restrict__ x, const int* __restrict__ ypast,
    const float* __restrict__ ln_g, const float* __restrict__ ln_b,
    const _Float16* __restrict__ w1h, const float* __restrict__ b1,
    const _Float16* __restrict__ w2h, const float* __restrict__ b2,
    const _Float16* __restrict__ w3h, const float* __restrict__ b3,
    const _Float16* __restrict__ w4h, const float* __restrict__ b4,
    const float* __restrict__ bn1_g, const float* __restrict__ bn1_b,
    const float* __restrict__ bn1_m, const float* __restrict__ bn1_v,
    const float* __restrict__ bn2_g, const float* __restrict__ bn2_b,
    const float* __restrict__ bn2_m, const float* __restrict__ bn2_v,
    _Float16* __restrict__ seq) {
  __shared__ _Float16 h0[16][512];
  __shared__ _Float16 h1[16][128];
  __shared__ _Float16 h2[16][64];
  __shared__ _Float16 h3[16][32];
  __shared__ float rs[16][8], rq[16][8], mu_s[16], rstd_s[16];

  const int tid = threadIdx.x;
  const int r0 = blockIdx.x * 16;
  const int row = tid >> 3, sub = tid & 7;

  // --- LayerNorm over F=512 (8 threads/row) ---
  const float* xr = x + (size_t)(r0 + row) * FF + sub * 64;
  float s = 0.f, q = 0.f;
  for (int i = 0; i < 64; ++i) { float v = xr[i]; s += v; q += v * v; }
  rs[row][sub] = s; rq[row][sub] = q;
  __syncthreads();
  if (sub == 0) {
    float ss = 0.f, qq = 0.f;
    for (int i = 0; i < 8; ++i) { ss += rs[row][i]; qq += rq[row][i]; }
    float mu = ss / FF;
    mu_s[row] = mu;
    rstd_s[row] = rsqrtf(qq / FF - mu * mu + EPSF);
  }
  __syncthreads();
  {
    float mu = mu_s[row], rstd = rstd_s[row];
    for (int i = 0; i < 64; ++i) {
      int c = sub * 64 + i;
      h0[row][c] = (_Float16)(((xr[i] - mu) * rstd) * ln_g[c] + ln_b[c]);
    }
  }
  __syncthreads();

  const int w = tid >> 5;                 // wave id (uniform per wave)
  const int lane = tid & 31;
  const int col16 = lane & 15, rbase = (lane >> 4) * 8;

  // --- layer1 [16,512]@[512,128] + BN1, 2 col-tiles per wave ---
#pragma unroll
  for (int jt = 0; jt < 2; ++jt) {
    int nt = w * 2 + jt;
    v8f acc = {};
#pragma unroll
    for (int kc = 0; kc < 16; ++kc) {
      v16h a = load_a16(&h0[0][kc * 32], 512);
      v16h b = load_b_swz(w1h + ((size_t)(kc * 8 + nt)) * 512);
      acc = wmma_f16(a, b, acc);
    }
    int col = nt * 16 + col16;
    float g  = bn1_g[col] * rsqrtf(bn1_v[col] + EPSF);
    float bb = bn1_b[col] - bn1_m[col] * g;
    float bias = b1[col];
#pragma unroll
    for (int i = 0; i < 8; ++i) {
      float v = acc[i] + bias; v = v > 0.f ? v : 0.f;
      h1[rbase + i][col] = (_Float16)(v * g + bb);
    }
  }
  __syncthreads();

  // --- layer2 [16,128]@[128,64], 1 col-tile per wave ---
  {
    int nt = w;
    v8f acc = {};
#pragma unroll
    for (int kc = 0; kc < 4; ++kc) {
      v16h a = load_a16(&h1[0][kc * 32], 128);
      v16h b = load_b_swz(w2h + ((size_t)(kc * 4 + nt)) * 512);
      acc = wmma_f16(a, b, acc);
    }
    int col = nt * 16 + col16;
    float bias = b2[col];
#pragma unroll
    for (int i = 0; i < 8; ++i) {
      float v = acc[i] + bias;
      h2[rbase + i][col] = (_Float16)(v > 0.f ? v : 0.f);
    }
  }
  __syncthreads();

  // --- layer3 [16,64]@[64,32], waves 0..1 ---
  if (w < 2) {
    int nt = w;
    v8f acc = {};
#pragma unroll
    for (int kc = 0; kc < 2; ++kc) {
      v16h a = load_a16(&h2[0][kc * 32], 64);
      v16h b = load_b_swz(w3h + ((size_t)(kc * 2 + nt)) * 512);
      acc = wmma_f16(a, b, acc);
    }
    int col = nt * 16 + col16;
    float bias = b3[col];
#pragma unroll
    for (int i = 0; i < 8; ++i) {
      float v = acc[i] + bias;
      h3[rbase + i][col] = (_Float16)(v > 0.f ? v : 0.f);
    }
  }
  __syncthreads();

  // --- layer4 [16,32]@[32,16] + BN2 -> seq cols 0..15 (wave 0) ---
  if (w == 0) {
    v8f acc = {};
    v16h a = load_a16(&h3[0][0], 32);
    v16h b = load_b_swz(w4h);
    acc = wmma_f16(a, b, acc);
    int col = col16;
    float g  = bn2_g[col] * rsqrtf(bn2_v[col] + EPSF);
    float bb = bn2_b[col] - bn2_m[col] * g;
    float bias = b4[col];
#pragma unroll
    for (int i = 0; i < 8; ++i) {
      float v = acc[i] + bias; v = v > 0.f ? v : 0.f;
      seq[(size_t)(r0 + rbase + i) * 32 + col] = (_Float16)(v * g + bb);
    }
  } else if (w == 1) {
    // cols 16..31: y_past then zero pad
    int idx = tid - 32;
    for (int j = idx; j < 256; j += 32) {
      int rr = j >> 4, cc = 16 + (j & 15);
      _Float16 v = (_Float16)0.0f;
      if (cc == 16) v = (_Float16)(float)ypast[r0 + rr];
      seq[(size_t)(r0 + rr) * 32 + cc] = v;
    }
  }
}

// ---------------------------------------------------------------------------
// One LSTM timestep, both directions. Grid (4, 32, 2), block = 128 (4 waves).
// Block owns one 16-col N-tile; each wave a different 16-row batch tile.
// Weight fragments for the block's N-tile (17 K-chunks x 4 gates) are staged
// through LDS in two phases (36KB buffer) so all 4 waves share one L2 read.
// K = 32(seq, padded 17) + 512(h) = 17 chunks; 68 v_wmma per wave.
// ---------------------------------------------------------------------------
__global__ __launch_bounds__(128) void k_lstm_step(
    const _Float16* __restrict__ seq,
    const _Float16* __restrict__ encW0, const _Float16* __restrict__ encW1,
    const float* __restrict__ bias0, const float* __restrict__ bias1,
    const _Float16* __restrict__ hin0, const _Float16* __restrict__ hin1,
    _Float16* __restrict__ hout0, _Float16* __restrict__ hout1,
    float* __restrict__ c0, float* __restrict__ c1,
    _Float16* __restrict__ hs0, _Float16* __restrict__ hs1, int t) {
  __shared__ _Float16 bs[9 * 4 * 512];   // 36KB fragment stage
  const int d = blockIdx.z;
  const _Float16* encW = d ? encW1 : encW0;
  const float*    bias = d ? bias1 : bias0;
  const _Float16* hin  = d ? hin1 : hin0;
  _Float16*       hout = d ? hout1 : hout0;
  float*          cbuf = d ? c1 : c0;
  _Float16*       hs   = d ? hs1 : hs0;
  const int td = d ? (TT - 1 - t) : t;
  const int n0 = blockIdx.y * 16;
  const int ncol = n0 >> 4;               // column tile within a gate block
  const int w = threadIdx.x >> 5;
  const int b0 = (blockIdx.x * 4 + w) * 16;

  v8f acc0 = {}, acc1 = {}, acc2 = {}, acc3 = {};
#pragma unroll
  for (int ph = 0; ph < 2; ++ph) {
    const int kbeg = ph ? 9 : 0;
    const int kcnt = ph ? 8 : 9;
    // --- stage kcnt*4 fragments (each 512 halves) into LDS ---
    {
      const int total = kcnt * 4 * 512;   // halves; divisible by 128*8
      for (int e = threadIdx.x * 8; e < total; e += 128 * 8) {
        int frag = e >> 9;                // krel*4 + gate
        int off  = e & 511;
        int krel = frag >> 2, gate = frag & 3;
        const _Float16* src =
            encW + ((size_t)((kbeg + krel) * 128 + gate * 32 + ncol)) * 512 + off;
        stage16(src, &bs[frag * 512 + off]);
      }
    }
    wait_async_and_bar();
    // --- consume ---
#pragma unroll
    for (int kr = 0; kr < 9; ++kr) {
      if (kr >= kcnt) break;
      const int k = kbeg + kr;
      v16h a = (k == 0)
                   ? load_a16(seq + ((size_t)b0 * TT + td) * 32, TT * 32)
                   : load_a16(hin + (size_t)b0 * HH + (k - 1) * 32, HH);
      const _Float16* bf = &bs[(kr * 4) * 512];
      v16h b;
      b = load_b_swz(bf);            acc0 = wmma_f16(a, b, acc0);
      b = load_b_swz(bf + 512);      acc1 = wmma_f16(a, b, acc1);
      b = load_b_swz(bf + 1024);     acc2 = wmma_f16(a, b, acc2);
      b = load_b_swz(bf + 1536);     acc3 = wmma_f16(a, b, acc3);
    }
    __syncthreads();                 // protect bs before next phase overwrite
  }

  const int lane = threadIdx.x & 31;
  const int col = n0 + (lane & 15);
  const int rbase = b0 + (lane >> 4) * 8;
  const float bi = bias[col], bf_ = bias[512 + col], bg = bias[1024 + col],
              bo = bias[1536 + col];
#pragma unroll
  for (int i = 0; i < 8; ++i) {
    int m = rbase + i;
    float zi = acc0[i] + bi, zf = acc1[i] + bf_, zg = acc2[i] + bg,
          zo = acc3[i] + bo;
    float cold = cbuf[(size_t)m * HH + col];
    float cn = sigm(zf) * cold + sigm(zi) * tanhf(zg);
    float hn = sigm(zo) * tanhf(cn);
    cbuf[(size_t)m * HH + col] = cn;
    hout[(size_t)m * HH + col] = (_Float16)hn;
    hs[((size_t)m * TT + td) * HH + col] = (_Float16)hn;
  }
}

// ---------------------------------------------------------------------------
// Projection: enc = concat(hs_f[t], hs_b[t]) @ proj_w + proj_b  -> f32 [B,T,H]
// (hs_b is already stored indexed by original time, so no reversal needed.)
// Grid (4096, 8), block 128: block = 16 rows x 64 cols, wave = 16x16, K=1024.
// ---------------------------------------------------------------------------
__global__ __launch_bounds__(128) void k_proj(
    const _Float16* __restrict__ hsf, const _Float16* __restrict__ hsb,
    const _Float16* __restrict__ pw, const float* __restrict__ pb,
    float* __restrict__ encp) {
  const int r0 = blockIdx.x * 16;
  const int w = threadIdx.x >> 5;
  const int n0 = blockIdx.y * 64 + w * 16;
  const int ncol = n0 >> 4;
  v8f acc = {};
  const _Float16* Af = hsf + (size_t)r0 * HH;   // rows share batch (256 % 16)
  const _Float16* Ab = hsb + (size_t)r0 * HH;
#pragma unroll 4
  for (int kc = 0; kc < 16; ++kc) {
    v16h a = load_a16(Af + kc * 32, HH);
    v16h b = load_b_swz(pw + ((size_t)kc * 32 + ncol) * 512);
    acc = wmma_f16(a, b, acc);
  }
#pragma unroll 4
  for (int kc = 0; kc < 16; ++kc) {
    v16h a = load_a16(Ab + kc * 32, HH);
    v16h b = load_b_swz(pw + ((size_t)(16 + kc) * 32 + ncol) * 512);
    acc = wmma_f16(a, b, acc);
  }
  const int lane = threadIdx.x & 31;
  const int col = n0 + (lane & 15), rbase = (lane >> 4) * 8;
  const float bias = pb[col];
#pragma unroll
  for (int i = 0; i < 8; ++i)
    encp[(size_t)(r0 + rbase + i) * HH + col] = acc[i] + bias;
}

// ---------------------------------------------------------------------------
// state_h / state_c: concat(hT_f, hT_b) @ sph_w (+b), concat(cT_f, cT_b) @ spc_w.
// Grid (16, 8, 2), block 128.
// ---------------------------------------------------------------------------
__global__ __launch_bounds__(128) void k_state(
    const _Float16* __restrict__ hf, const _Float16* __restrict__ hb,
    const float* __restrict__ cf, const float* __restrict__ cb,
    const _Float16* __restrict__ sphW, const float* __restrict__ sphB,
    const _Float16* __restrict__ spcW, const float* __restrict__ spcB,
    float* __restrict__ state_h, _Float16* __restrict__ state_h16,
    float* __restrict__ state_c) {
  const int which = blockIdx.z;
  const int r0 = blockIdx.x * 16;
  const int w = threadIdx.x >> 5;
  const int n0 = blockIdx.y * 64 + w * 16;
  const int ncol = n0 >> 4;
  const _Float16* W = which ? spcW : sphW;
  const float* bias = which ? spcB : sphB;
  v8f acc = {};
  if (which == 0) {
#pragma unroll 4
    for (int kc = 0; kc < 16; ++kc) {
      v16h a = load_a16(hf + (size_t)r0 * HH + kc * 32, HH);
      v16h b = load_b_swz(W + ((size_t)kc * 32 + ncol) * 512);
      acc = wmma_f16(a, b, acc);
    }
#pragma unroll 4
    for (int kc = 0; kc < 16; ++kc) {
      v16h a = load_a16(hb + (size_t)r0 * HH + kc * 32, HH);
      v16h b = load_b_swz(W + ((size_t)(16 + kc) * 32 + ncol) * 512);
      acc = wmma_f16(a, b, acc);
    }
  } else {
#pragma unroll 4
    for (int kc = 0; kc < 16; ++kc) {
      v16h a = load_a16_f32(cf + (size_t)r0 * HH + kc * 32, HH);
      v16h b = load_b_swz(W + ((size_t)kc * 32 + ncol) * 512);
      acc = wmma_f16(a, b, acc);
    }
#pragma unroll 4
    for (int kc = 0; kc < 16; ++kc) {
      v16h a = load_a16_f32(cb + (size_t)r0 * HH + kc * 32, HH);
      v16h b = load_b_swz(W + ((size_t)(16 + kc) * 32 + ncol) * 512);
      acc = wmma_f16(a, b, acc);
    }
  }
  const int lane = threadIdx.x & 31;
  const int col = n0 + (lane & 15), rbase = (lane >> 4) * 8;
  const float bv = bias[col];
#pragma unroll
  for (int i = 0; i < 8; ++i) {
    float v = acc[i] + bv;
    size_t idx = (size_t)(r0 + rbase + i) * HH + col;
    if (which == 0) { state_h[idx] = v; state_h16[idx] = (_Float16)v; }
    else            { state_c[idx] = v; }
  }
}

// ---------------------------------------------------------------------------
// Dot-product attention per batch: scores (T=256), softmax, weighted sum.
// Grid 256, block 256.
// ---------------------------------------------------------------------------
__global__ __launch_bounds__(256) void k_attn(
    const float* __restrict__ state_h, const float* __restrict__ encp,
    _Float16* __restrict__ attn16) {
  __shared__ float sh[512];
  __shared__ float red[256];
  __shared__ float wgt[256];
  const int b = blockIdx.x, tid = threadIdx.x;
  sh[tid]       = state_h[(size_t)b * HH + tid];
  sh[tid + 256] = state_h[(size_t)b * HH + tid + 256];
  __syncthreads();
  const float* er = encp + ((size_t)b * TT + tid) * HH;
  float s = 0.f;
  for (int k = 0; k < 512; ++k) s += sh[k] * er[k];
  red[tid] = s; __syncthreads();
  for (int off = 128; off > 0; off >>= 1) {
    if (tid < off) red[tid] = fmaxf(red[tid], red[tid + off]);
    __syncthreads();
  }
  float mx = red[0]; __syncthreads();
  float e = __expf(s - mx);
  red[tid] = e; __syncthreads();
  for (int off = 128; off > 0; off >>= 1) {
    if (tid < off) red[tid] += red[tid + off];
    __syncthreads();
  }
  float sum = red[0]; __syncthreads();
  wgt[tid] = e / sum; __syncthreads();
  for (int c = tid; c < 512; c += 256) {
    float a = 0.f;
    for (int t = 0; t < 256; ++t)
      a += wgt[t] * encp[((size_t)b * TT + t) * HH + c];
    attn16[(size_t)b * HH + c] = (_Float16)a;
  }
}

// ---------------------------------------------------------------------------
// Decoder LSTM cell: z = [attn | state_h] @ [dec_wx; dec_wh] + bias; gates.
// Grid (16, 32), block = 1 wave; K = 1024 -> 32 chunks, 128 v_wmma per wave.
// ---------------------------------------------------------------------------
__global__ __launch_bounds__(32) void k_dec(
    const _Float16* __restrict__ attn16, const _Float16* __restrict__ sh16,
    const _Float16* __restrict__ decW, const float* __restrict__ dbias,
    const float* __restrict__ state_c, _Float16* __restrict__ hdec16) {
  const int b0 = blockIdx.x * 16, n0 = blockIdx.y * 16;
  const int ncol = n0 >> 4;
  v8f acc0 = {}, acc1 = {}, acc2 = {}, acc3 = {};
#pragma unroll 4
  for (int k = 0; k < 32; ++k) {
    const _Float16* Ap = (k < 16) ? (attn16 + (size_t)b0 * HH + k * 32)
                                  : (sh16 + (size_t)b0 * HH + (k - 16) * 32);
    v16h a = load_a16(Ap, HH);
    const _Float16* wf = decW + ((size_t)k * 128 + ncol) * 512;
    v16h b;
    b = load_b_swz(wf);                    acc0 = wmma_f16(a, b, acc0);
    b = load_b_swz(wf + (size_t)32 * 512); acc1 = wmma_f16(a, b, acc1);
    b = load_b_swz(wf + (size_t)64 * 512); acc2 = wmma_f16(a, b, acc2);
    b = load_b_swz(wf + (size_t)96 * 512); acc3 = wmma_f16(a, b, acc3);
  }
  const int lane = threadIdx.x & 31;
  const int col = n0 + (lane & 15), rbase = b0 + (lane >> 4) * 8;
  const float bi = dbias[col], bf = dbias[512 + col], bg = dbias[1024 + col],
              bo = dbias[1536 + col];
#pragma unroll
  for (int i = 0; i < 8; ++i) {
    int m = rbase + i;
    float zi = acc0[i] + bi, zf = acc1[i] + bf, zg = acc2[i] + bg,
          zo = acc3[i] + bo;
    float cold = state_c[(size_t)m * HH + col];
    float cn = sigm(zf) * cold + sigm(zi) * tanhf(zg);
    float hn = sigm(zo) * tanhf(cn);
    hdec16[(size_t)m * HH + col] = (_Float16)hn;
  }
}

// Output head: [256,512] @ [512,2] + b -> d_out f32 [256,2].
__global__ __launch_bounds__(256) void k_out(
    const _Float16* __restrict__ hdec16, const float* __restrict__ ow,
    const float* __restrict__ ob, float* __restrict__ out) {
  int id = blockIdx.x * blockDim.x + threadIdx.x;
  if (id >= BB * 2) return;
  int b = id >> 1, j = id & 1;
  float s = ob[j];
  for (int k = 0; k < HH; ++k)
    s += (float)hdec16[(size_t)b * HH + k] * ow[k * 2 + j];
  out[id] = s;
}

// ---------------------------------------------------------------------------
extern "C" void kernel_launch(void* const* d_in, const int* in_sizes, int n_in,
                              void* d_out, int out_size, void* d_ws,
                              size_t ws_size, hipStream_t stream) {
  const float* x      = (const float*)d_in[0];
  const int*   y_past = (const int*)d_in[1];
  const float* ln_g = (const float*)d_in[2];
  const float* ln_b = (const float*)d_in[3];
  const float* w1 = (const float*)d_in[4];  const float* b1 = (const float*)d_in[5];
  const float* w2 = (const float*)d_in[6];  const float* b2 = (const float*)d_in[7];
  const float* w3 = (const float*)d_in[8];  const float* b3 = (const float*)d_in[9];
  const float* w4 = (const float*)d_in[10]; const float* b4 = (const float*)d_in[11];
  const float* bn1_g = (const float*)d_in[12]; const float* bn1_b = (const float*)d_in[13];
  const float* bn1_m = (const float*)d_in[14]; const float* bn1_v = (const float*)d_in[15];
  const float* bn2_g = (const float*)d_in[16]; const float* bn2_b = (const float*)d_in[17];
  const float* bn2_m = (const float*)d_in[18]; const float* bn2_v = (const float*)d_in[19];
  const float* efwx = (const float*)d_in[20]; const float* efwh = (const float*)d_in[21];
  const float* efb  = (const float*)d_in[22];
  const float* ebwx = (const float*)d_in[23]; const float* ebwh = (const float*)d_in[24];
  const float* ebb  = (const float*)d_in[25];
  const float* dwx = (const float*)d_in[26]; const float* dwh = (const float*)d_in[27];
  const float* dbias = (const float*)d_in[28];
  const float* pw = (const float*)d_in[29];  const float* pb = (const float*)d_in[30];
  const float* sphw = (const float*)d_in[31]; const float* sphb = (const float*)d_in[32];
  const float* spcw = (const float*)d_in[33]; const float* spcb = (const float*)d_in[34];
  const float* ow = (const float*)d_in[35];  const float* obias = (const float*)d_in[36];
  float* out = (float*)d_out;

  // Workspace bump allocator (256B aligned).
  char* ws = (char*)d_ws;
  size_t off = 0;
  auto alloc = [&](size_t bytes) -> void* {
    void* p = ws + off;
    off = (off + bytes + 255) & ~(size_t)255;
    return p;
  };
  _Float16* seq    = (_Float16*)alloc((size_t)BB * TT * 32 * 2);
  _Float16* w1h    = (_Float16*)alloc((size_t)512 * 128 * 2);
  _Float16* w2h    = (_Float16*)alloc((size_t)128 * 64 * 2);
  _Float16* w3h    = (_Float16*)alloc((size_t)64 * 32 * 2);
  _Float16* w4h    = (_Float16*)alloc((size_t)32 * 16 * 2);
  _Float16* encW0  = (_Float16*)alloc((size_t)544 * 2048 * 2);
  _Float16* encW1  = (_Float16*)alloc((size_t)544 * 2048 * 2);
  _Float16* decW   = (_Float16*)alloc((size_t)1024 * 2048 * 2);
  _Float16* projWh = (_Float16*)alloc((size_t)1024 * 512 * 2);
  _Float16* sphWh  = (_Float16*)alloc((size_t)1024 * 512 * 2);
  _Float16* spcWh  = (_Float16*)alloc((size_t)1024 * 512 * 2);
  _Float16* h0p0   = (_Float16*)alloc((size_t)BB * HH * 2);
  _Float16* h0p1   = (_Float16*)alloc((size_t)BB * HH * 2);
  _Float16* h1p0   = (_Float16*)alloc((size_t)BB * HH * 2);
  _Float16* h1p1   = (_Float16*)alloc((size_t)BB * HH * 2);
  float*    cbuf0  = (float*)alloc((size_t)BB * HH * 4);
  float*    cbuf1  = (float*)alloc((size_t)BB * HH * 4);
  _Float16* hs0    = (_Float16*)alloc((size_t)BB * TT * HH * 2);
  _Float16* hs1    = (_Float16*)alloc((size_t)BB * TT * HH * 2);
  float*    encp   = (float*)alloc((size_t)BB * TT * HH * 4);
  float*    stateh = (float*)alloc((size_t)BB * HH * 4);
  _Float16* sh16   = (_Float16*)alloc((size_t)BB * HH * 2);
  float*    statec = (float*)alloc((size_t)BB * HH * 4);
  _Float16* attn16 = (_Float16*)alloc((size_t)BB * HH * 2);
  _Float16* hdec16 = (_Float16*)alloc((size_t)BB * HH * 2);
  (void)ws_size; (void)in_sizes; (void)n_in; (void)out_size;

  auto pack = [&](_Float16* dst, const float* src, int K, int N) {
    int n = K * N;
    k_pack<<<(n + 255) / 256, 256, 0, stream>>>(dst, src, K, N);
  };
  // Pack weights into WMMA B-fragment layout (f16).
  pack(w1h, w1, 512, 128);
  pack(w2h, w2, 128, 64);
  pack(w3h, w3, 64, 32);
  pack(w4h, w4, 32, 16);
  pack(projWh, pw, 1024, 512);
  pack(sphWh, sphw, 1024, 512);
  pack(spcWh, spcw, 1024, 512);
  {
    int n = 544 * 2048;
    k_pack_lstm<<<(n + 255) / 256, 256, 0, stream>>>(encW0, efwx, efwh);
    k_pack_lstm<<<(n + 255) / 256, 256, 0, stream>>>(encW1, ebwx, ebwh);
    int m = 1024 * 2048;
    k_pack_dec<<<(m + 255) / 256, 256, 0, stream>>>(decW, dwx, dwh);
  }
  // Zero initial recurrent state (ws is poisoned; must re-init every call).
  {
    int n = BB * HH;
    k_zero_h<<<(n + 255) / 256, 256, 0, stream>>>(h0p0, n);
    k_zero_h<<<(n + 255) / 256, 256, 0, stream>>>(h1p0, n);
    k_zero_f<<<(n + 255) / 256, 256, 0, stream>>>(cbuf0, n);
    k_zero_f<<<(n + 255) / 256, 256, 0, stream>>>(cbuf1, n);
  }
  // LayerNorm + MLP + BN -> seq f16 [B*T, 32]
  k_mlp<<<(BB * TT) / 16, 128, 0, stream>>>(
      x, y_past, ln_g, ln_b, w1h, b1, w2h, b2, w3h, b3, w4h, b4,
      bn1_g, bn1_b, bn1_m, bn1_v, bn2_g, bn2_b, bn2_m, bn2_v, seq);
  // Bidirectional LSTM: 256 steps, fwd+bwd fused. h ping-pongs parity.
  for (int t = 0; t < TT; ++t) {
    const _Float16* hi0 = (t & 1) ? h0p1 : h0p0;
    const _Float16* hi1 = (t & 1) ? h1p1 : h1p0;
    _Float16* ho0 = (t & 1) ? h0p0 : h0p1;
    _Float16* ho1 = (t & 1) ? h1p0 : h1p1;
    k_lstm_step<<<dim3(4, 32, 2), 128, 0, stream>>>(
        seq, encW0, encW1, efb, ebb, hi0, hi1, ho0, ho1, cbuf0, cbuf1,
        hs0, hs1, t);
  }
  // Final states land in parity 0 (t=255 writes p0).
  // Projection to enc [B,T,H] f32.
  k_proj<<<dim3((BB * TT) / 16, 8), 128, 0, stream>>>(hs0, hs1, projWh, pb, encp);
  // state_h / state_c heads.
  k_state<<<dim3(16, 8, 2), 128, 0, stream>>>(h0p0, h1p0, cbuf0, cbuf1, sphWh,
                                              sphb, spcWh, spcb, stateh, sh16,
                                              statec);
  // Attention.
  k_attn<<<BB, 256, 0, stream>>>(stateh, encp, attn16);
  // Decoder cell.
  k_dec<<<dim3(16, 32), 32, 0, stream>>>(attn16, sh16, decW, dbias, statec,
                                         hdec16);
  // Output head.
  k_out<<<2, 256, 0, stream>>>(hdec16, ow, obias, out);
}
